// Wave_UGNN_50903952392326
// MI455X (gfx1250) — matrix operand order repaired and checked by reading, not compiled, not run on hardware
//
#include <hip/hip_runtime.h>
#include <hip/hip_bf16.h>

typedef __attribute__((ext_vector_type(16))) _Float16 v16h;
typedef __attribute__((ext_vector_type(8)))  _Float16 v8h;
typedef __attribute__((ext_vector_type(8)))  float    v8f;

#define LEAK_F 0.2f

// ---------------------------------------------------------------- utilities
__global__ void f32_to_f16_kernel(const float* __restrict__ src,
                                  _Float16* __restrict__ dst, int n) {
  int i = blockIdx.x * blockDim.x + threadIdx.x;
  if (i < n) dst[i] = (_Float16)src[i];
}

// -------------------------------------------------- weight repack (one-time)
// Conv weights -> WMMA A-fragment order: [coTile][kChunk][lane][h] halves.
// A layout (16-bit 16x32): lane l holds M=l&15, K = (h&7) + 8*(l>=16) + 16*(h>=8)
__global__ void pack_conv_w_kernel(const float* __restrict__ w,
                                   _Float16* __restrict__ wp, int Co, int Cin,
                                   int taps, int ciPer, int nchunks, long total) {
  long idx = (long)blockIdx.x * blockDim.x + threadIdx.x;
  if (idx >= total) return;
  int e = (int)(idx & 511);
  long tc = idx >> 9;
  int chunk = (int)(tc % nchunks);
  int tile = (int)(tc / nchunks);
  int lane = e >> 4, h = e & 15;
  int ka = (h & 7) + ((lane >> 4) << 3) + ((h >> 3) << 4);
  int ci = chunk * ciPer + ka / taps;
  int kk = ka - (ka / taps) * taps;
  int co = tile * 16 + (lane & 15);
  float v = 0.f;
  if (ka < taps * ciPer && ci < Cin && co < Co)
    v = w[((long)co * Cin + ci) * taps + kk];
  wp[idx] = (_Float16)v;
}

// Dense B operand [K][N] -> WMMA B-fragment order: [nTile][kChunk][lane][h]
// B layout: lane l holds N=l&15, K = (h&7) + 8*(l>=16) + 16*(h>=8)
__global__ void pack_matB_kernel(const float* __restrict__ W,
                                 _Float16* __restrict__ wp, int K, int N,
                                 int nchunks, long total) {
  long idx = (long)blockIdx.x * blockDim.x + threadIdx.x;
  if (idx >= total) return;
  int e = (int)(idx & 511);
  long tc = idx >> 9;
  int chunk = (int)(tc % nchunks);
  int tile = (int)(tc / nchunks);
  int lane = e >> 4, h = e & 15;
  int n = tile * 16 + (lane & 15);
  int k = chunk * 32 + (h & 7) + ((lane >> 4) << 3) + ((h >> 3) << 4);
  float v = (k < K) ? W[(long)k * N + n] : 0.f;
  wp[idx] = (_Float16)v;
}

// ------------------------------------------------- conv1d as implicit GEMM
// Block = 128 threads (4 waves). All waves share one 16-co tile; wave wv owns
// positions [ltb+16*wv, +16). Input window staged in LDS per K-chunk
// (branchless fill: clamped address + select-zero; pad row stays zero).
// A fragments come pre-packed from global (2 x b128 per lane). Bias+Leaky fused.
template <int TAPS, int CI_PER>
__global__ __launch_bounds__(128) void conv1d_wmma_kernel(
    const _Float16* __restrict__ in1, int C1, int rs1, int step,
    const _Float16* __restrict__ in2, int C2, int rs2,
    const _Float16* __restrict__ wpack, const float* __restrict__ bias,
    _Float16* __restrict__ out, int Co, int Lout, int nchunks) {
  constexpr int PAD = TAPS / 2;
  constexpr int POSW = 64 + 2 * PAD;        // staged positions per channel
  constexpr int FILLN = CI_PER * POSW;      // refilled every chunk
  constexpr int LDSN = (CI_PER + 1) * POSW + 32;  // + zero pad row
  __shared__ _Float16 Blds[LDSN];

  const int tid = threadIdx.x;
  const int lane = tid & 31;
  const int wv = tid >> 5;
  const int ltb = blockIdx.x * 64;
  const int b = blockIdx.z;
  const int Cin = C1 + C2;
  const int mn = lane & 15;
  const int khi8 = (lane >> 4) << 3;
  const int khi16 = (lane >> 4) << 4;
  const long ib1 = (long)b * C1 * rs1;
  const long ib2 = (long)b * C2 * rs2;

  // zero the pad region once (rows >= CI_PER are never written by fills)
  for (int e = FILLN + tid; e < LDSN; e += 128) Blds[e] = (_Float16)0.0f;

  const _Float16* wp = wpack + ((long)blockIdx.y * nchunks) * 512 + lane * 16;

  v8f acc = {};
  for (int chunk = 0; chunk < nchunks; ++chunk) {
    const int ci0 = chunk * CI_PER;
    __syncthreads();  // previous compute done / pad zeros visible
    for (int e = tid; e < FILLN; e += 128) {
      int row = e / POSW;
      int p = e - row * POSW;
      int ci = ci0 + row;
      int pos = ltb - PAD + p;
      int posc = pos < 0 ? 0 : (pos >= Lout ? Lout - 1 : pos);
      int cic = ci < Cin ? ci : Cin - 1;
      const _Float16* src = (cic < C1)
          ? in1 + ib1 + (long)cic * rs1 + (long)posc * step
          : in2 + ib2 + (long)(cic - C1) * rs2 + posc;
      _Float16 v = *src;  // always in-bounds (clamped)
      long nextRow = (cic < C1) ? (long)CI_PER * rs1 : (long)CI_PER * rs2;
      __builtin_prefetch(src + nextRow, 0, 0);  // next K-chunk's row
      bool ok = (ci < Cin) & (pos >= 0) & (pos < Lout);
      Blds[e] = ok ? v : (_Float16)0.0f;
    }
    __syncthreads();
    // A fragment: packed weights, 2 contiguous b128 loads
    const v8h* ap = (const v8h*)(wp + (long)chunk * 512);
    v8h a0 = ap[0], a1 = ap[1];
    v16h afrag = __builtin_shufflevector(a0, a1, 0, 1, 2, 3, 4, 5, 6, 7, 8, 9,
                                         10, 11, 12, 13, 14, 15);
    // B fragment: unconditional LDS gathers (pad K lanes hit the zero row)
    v16h bfrag;
#pragma unroll
    for (int h = 0; h < 16; ++h) {
      int k = (h & 7) + khi8 + ((h >> 3) << 4);
      int ci = k / TAPS;
      int kk = k - ci * TAPS;
      bfrag[h] = Blds[ci * POSW + wv * 16 + mn + kk];
    }
    acc = __builtin_amdgcn_wmma_f32_16x16x32_f16(false, afrag, false, bfrag,
                                                 (short)0, acc, false, false);
  }
  // D layout: element r -> M = r + 8*(lane>=16), N = lane&15
  const int n = ltb + wv * 16 + mn;
  const int cot = blockIdx.y * 16;
#pragma unroll
  for (int r = 0; r < 8; ++r) {
    int coW = cot + r + khi8;
    if (coW < Co) {
      float v = acc[r] + bias[coW];
      v = v > 0.0f ? v : LEAK_F * v;
      out[((long)b * Co + coW) * Lout + n] = (_Float16)v;
    }
  }
}

// ---------------------------------------------------------- GNN dense GEMM
// out[N][Cout](f32) = A[N][256-padded](f16) @ Wpacked.  A tail cols are zero,
// packed W tail rows are zero, so the K loop is branch-free b128 loads.
__global__ __launch_bounds__(128) void matmul_wmma_kernel(
    const _Float16* __restrict__ A, const _Float16* __restrict__ wpack,
    float* __restrict__ out, int Cout, int nchunks) {
  const int lane = threadIdx.x & 31;
  const int wv = threadIdx.x >> 5;
  const int nt = (blockIdx.y * 4 + wv) * 16;
  const int mn = lane & 15;
  const int khi8 = (lane >> 4) << 3;
  const _Float16* arow = A + (long)(nt + mn) * 256;
  const _Float16* wp = wpack + ((long)blockIdx.x * nchunks) * 512 + lane * 16;
  v8f acc = {};
  for (int chunk = 0; chunk < nchunks; ++chunk) {
    int k0 = chunk * 32;
    v8h a0 = *(const v8h*)(arow + k0 + khi8);
    v8h a1 = *(const v8h*)(arow + k0 + khi8 + 16);
    v16h afrag = __builtin_shufflevector(a0, a1, 0, 1, 2, 3, 4, 5, 6, 7, 8, 9,
                                         10, 11, 12, 13, 14, 15);
    const v8h* bp = (const v8h*)(wp + (long)chunk * 512);
    v16h bfrag = __builtin_shufflevector(bp[0], bp[1], 0, 1, 2, 3, 4, 5, 6, 7,
                                         8, 9, 10, 11, 12, 13, 14, 15);
    acc = __builtin_amdgcn_wmma_f32_16x16x32_f16(false, afrag, false, bfrag,
                                                 (short)0, acc, false, false);
  }
  const int ct = blockIdx.x * 16;
#pragma unroll
  for (int r = 0; r < 8; ++r)
    out[(long)(nt + r + khi8) * Cout + ct + mn] = acc[r];
}

// ---------------------------------------- GCN chain aggregation (analytic)
__device__ __forceinline__ float dinv_of(int jl) {
  int a = jl < 4 ? jl : 4;
  int bd = 2047 - jl; int bb = bd < 4 ? bd : 4;
  return rsqrtf((float)(a + bb + 1));
}

__global__ void gcn_agg_kernel(const float* __restrict__ h,
                               const float* __restrict__ bias,
                               _Float16* __restrict__ outNodes,  // [8192][256] or null
                               _Float16* __restrict__ outBott,   // [4][C][2048] or null
                               int C, int doRelu) {
  long idx = (long)blockIdx.x * blockDim.x + threadIdx.x;
  long total = (long)8192 * C;
  if (idx >= total) return;
  int j = (int)(idx / C);
  int c = (int)(idx - (long)j * C);
  int jl = j & 2047;
  float dj = dinv_of(jl);
  float s = h[(long)j * C + c] * dj;
#pragma unroll
  for (int d = 1; d <= 4; ++d) {
    if (jl - d >= 0)    s += h[(long)(j - d) * C + c] * dinv_of(jl - d);
    if (jl + d <= 2047) s += h[(long)(j + d) * C + c] * dinv_of(jl + d);
  }
  float v = dj * s + bias[c];
  if (doRelu) v = v > 0.f ? v : 0.f;
  if (outNodes) outNodes[(long)j * 256 + c] = (_Float16)v;
  if (outBott) {
    int b = j >> 11, l = j & 2047;
    outBott[((long)b * C + c) * 2048 + l] = (_Float16)v;
  }
}

// skip5[B][144][4096] sampled ::2 -> node-major [8192][256], tail cols zeroed
__global__ void gather_nodes_kernel(const _Float16* __restrict__ s5,
                                    _Float16* __restrict__ nodes) {
  int idx = blockIdx.x * blockDim.x + threadIdx.x;
  if (idx >= 8192 * 256) return;
  int j = idx >> 8, c = idx & 255;
  int b = j >> 11, l = j & 2047;
  _Float16 v = (_Float16)0.0f;
  if (c < 144) v = s5[((long)b * 144 + c) * 4096 + 2 * l];
  nodes[idx] = v;
}

// Linear x2 upsample: pos = lo*(L-1)/(2L-1)
__global__ void upsample2_kernel(const _Float16* __restrict__ in,
                                 _Float16* __restrict__ out, int C, int L) {
  long idx = (long)blockIdx.x * blockDim.x + threadIdx.x;
  int Lo = 2 * L;
  long total = (long)4 * C * Lo;
  if (idx >= total) return;
  int lo = (int)(idx % Lo);
  long bc = idx / Lo;
  float pos = lo * ((float)(L - 1) / (float)(Lo - 1));
  int i0 = (int)floorf(pos);
  int i1 = i0 + 1 < L ? i0 + 1 : L - 1;
  float wgt = pos - (float)i0;
  float v0 = (float)in[bc * L + i0];
  float v1 = (float)in[bc * L + i1];
  out[bc * Lo + lo] = (_Float16)(v0 * (1.f - wgt) + v1 * wgt);
}

// final 1x1 conv over 24 channels + tanh -> f32 output
__global__ void outconv_tanh_kernel(const _Float16* __restrict__ in,
                                    const float* __restrict__ w,
                                    const float* __restrict__ ob,
                                    float* __restrict__ out) {
  const int C = 24; const long L = 131072;
  long idx = (long)blockIdx.x * blockDim.x + threadIdx.x;
  if (idx >= 4 * L) return;
  long b = idx / L, l = idx - b * L;
  float s = ob[0];
#pragma unroll
  for (int c = 0; c < C; ++c) s += w[c] * (float)in[((long)b * C + c) * L + l];
  out[idx] = tanhf(s);
}

// ------------------------------------------------------------------- host
extern "C" void kernel_launch(void* const* d_in, const int* in_sizes, int n_in,
                              void* d_out, int out_size, void* d_ws, size_t ws_size,
                              hipStream_t stream) {
  (void)in_sizes; (void)n_in; (void)out_size; (void)ws_size;
  const float* x = (const float*)d_in[0];
  const float *enc_w[6], *enc_b[6], *dec_w[6], *dec_b[6];
  for (int i = 0; i < 6; ++i) {
    enc_w[i] = (const float*)d_in[1 + i];
    enc_b[i] = (const float*)d_in[7 + i];
    dec_w[i] = (const float*)d_in[13 + i];
    dec_b[i] = (const float*)d_in[19 + i];
  }
  const float* gw1 = (const float*)d_in[25]; const float* gb1 = (const float*)d_in[26];
  const float* gw2 = (const float*)d_in[27]; const float* gb2 = (const float*)d_in[28];
  const float* gw3 = (const float*)d_in[29]; const float* gb3 = (const float*)d_in[30];
  const float* ow  = (const float*)d_in[31]; const float* obi = (const float*)d_in[32];
  // d_in[33]/[34] = edge_src/edge_dst: chain structure handled analytically

  static const int  encC[7]    = {1, 24, 48, 72, 96, 120, 144};
  static const long convL[6]   = {131072, 65536, 32768, 16384, 8192, 4096};
  static const int  encNch[6]  = {1, 12, 24, 36, 48, 60};   // ceil(Cin/2)
  static const int  encNt[6]   = {2, 3, 5, 6, 8, 9};        // ceil(Co/16)
  static const int  decCin[6]  = {288, 264, 216, 168, 120, 72};
  static const int  decCo[6]   = {144, 120, 96, 72, 48, 24};
  static const long decL[6]    = {4096, 8192, 16384, 32768, 65536, 131072};
  static const int  decNch[6]  = {48, 44, 36, 28, 20, 12};  // Cin/6
  static const int  decNt[6]   = {9, 8, 6, 5, 3, 2};

  char* ws = (char*)d_ws;
  size_t off = 0;
  auto alloc = [&](size_t elems, size_t esz) -> void* {
    void* p = ws + off;
    off += (elems * esz + 255) & ~(size_t)255;
    return p;
  };

  _Float16* xf = (_Float16*)alloc(524288, 2);
  _Float16 *wpEnc[6], *wpDec[6], *skip[6];
  long encPsz[6], decPsz[6];
  for (int i = 0; i < 6; ++i) {
    encPsz[i] = (long)encNt[i] * encNch[i] * 512;
    wpEnc[i] = (_Float16*)alloc(encPsz[i], 2);
  }
  for (int i = 0; i < 6; ++i) {
    decPsz[i] = (long)decNt[i] * decNch[i] * 512;
    wpDec[i] = (_Float16*)alloc(decPsz[i], 2);
  }
  _Float16* wpG1 = (_Float16*)alloc(16L * 5 * 512, 2);
  _Float16* wpG2 = (_Float16*)alloc(16L * 8 * 512, 2);
  _Float16* wpG3 = (_Float16*)alloc(9L * 8 * 512, 2);
  for (int i = 0; i < 6; ++i)
    skip[i] = (_Float16*)alloc(4L * encC[i + 1] * convL[i], 2);
  _Float16* nodes = (_Float16*)alloc(8192L * 256, 2);
  float*    htmp  = (float*)alloc(8192L * 256, 4);
  _Float16* bott  = (_Float16*)alloc(4L * 144 * 2048, 2);
  _Float16* up    = (_Float16*)alloc(25165824L, 2);   // max 4*48*131072
  _Float16* dout  = (_Float16*)alloc(12582912L, 2);   // max 4*24*131072

  // -------- one-time conversions / repacks
  f32_to_f16_kernel<<<(524288 + 255) / 256, 256, 0, stream>>>(x, xf, 524288);
  for (int i = 0; i < 6; ++i)
    pack_conv_w_kernel<<<(unsigned)((encPsz[i] + 255) / 256), 256, 0, stream>>>(
        enc_w[i], wpEnc[i], encC[i + 1], encC[i], 15, 2, encNch[i], encPsz[i]);
  for (int i = 0; i < 6; ++i)
    pack_conv_w_kernel<<<(unsigned)((decPsz[i] + 255) / 256), 256, 0, stream>>>(
        dec_w[i], wpDec[i], decCo[i], decCin[i], 5, 6, decNch[i], decPsz[i]);
  pack_matB_kernel<<<(16 * 5 * 512 + 255) / 256, 256, 0, stream>>>(
      gw1, wpG1, 144, 256, 5, 16L * 5 * 512);
  pack_matB_kernel<<<(16 * 8 * 512 + 255) / 256, 256, 0, stream>>>(
      gw2, wpG2, 256, 256, 8, 16L * 8 * 512);
  pack_matB_kernel<<<(9 * 8 * 512 + 255) / 256, 256, 0, stream>>>(
      gw3, wpG3, 256, 144, 8, 9L * 8 * 512);

  // -------- encoder (decimation folded into next layer's step=2)
  for (int i = 0; i < 6; ++i) {
    int Cin = encC[i], Co = encC[i + 1];
    long L = convL[i];
    const _Float16* in1 = (i == 0) ? xf : skip[i - 1];
    int rs = (i == 0) ? 131072 : (int)convL[i - 1];
    int step = (i == 0) ? 1 : 2;
    dim3 g((unsigned)(L / 64), (unsigned)((Co + 15) / 16), 4);
    conv1d_wmma_kernel<15, 2><<<g, 128, 0, stream>>>(
        in1, Cin, rs, step, (const _Float16*)nullptr, 0, 0,
        wpEnc[i], enc_b[i], skip[i], Co, (int)L, encNch[i]);
  }

  // -------- GNN bottleneck (8192 nodes, 3 GCN layers)
  gather_nodes_kernel<<<(8192 * 256 + 255) / 256, 256, 0, stream>>>(skip[5], nodes);
  {
    dim3 g1(16, 8192 / 64);
    matmul_wmma_kernel<<<g1, 128, 0, stream>>>(nodes, wpG1, htmp, 256, 5);
    gcn_agg_kernel<<<(8192L * 256 + 255) / 256, 256, 0, stream>>>(
        htmp, gb1, nodes, (_Float16*)nullptr, 256, 1);
    matmul_wmma_kernel<<<g1, 128, 0, stream>>>(nodes, wpG2, htmp, 256, 8);
    gcn_agg_kernel<<<(8192L * 256 + 255) / 256, 256, 0, stream>>>(
        htmp, gb2, nodes, (_Float16*)nullptr, 256, 1);
    dim3 g3(9, 8192 / 64);
    matmul_wmma_kernel<<<g3, 128, 0, stream>>>(nodes, wpG3, htmp, 144, 8);
    gcn_agg_kernel<<<(8192L * 144 + 255) / 256, 256, 0, stream>>>(
        htmp, gb3, (_Float16*)nullptr, bott, 144, 0);
  }

  // -------- decoder
  upsample2_kernel<<<(4L * 144 * 4096 + 255) / 256, 256, 0, stream>>>(bott, up, 144, 2048);
  for (int i = 0; i < 6; ++i) {
    if (i > 0) {
      long Lp = decL[i - 1];
      int Cp = decCo[i - 1];
      upsample2_kernel<<<(unsigned)((4L * Cp * 2 * Lp + 255) / 256), 256, 0, stream>>>(
          dout, up, Cp, (int)Lp);
    }
    int Cup = (i == 0) ? 144 : decCo[i - 1];
    int Cskip = encC[6 - i];
    long L = decL[i];
    dim3 g((unsigned)(L / 64), (unsigned)((decCo[i] + 15) / 16), 4);
    conv1d_wmma_kernel<5, 6><<<g, 128, 0, stream>>>(
        up, Cup, (int)L, 1, skip[5 - i], Cskip, (int)L,
        wpDec[i], dec_b[i], dout, decCo[i], (int)L, decNch[i]);
  }

  // -------- 1x1 output conv + tanh
  outconv_tanh_kernel<<<(524288 + 255) / 256, 256, 0, stream>>>(
      dout, ow, obi, (float*)d_out);
}
